// MultiHeadLinearAttention_67319317398040
// MI455X (gfx1250) — compile-verified, matrix operand order checked by reading, not run on hardware
//
#include <hip/hip_runtime.h>

// ---------------------------------------------------------------------------
// MultiHeadLinearAttention for MI455X (gfx1250), bf16 WMMA path.
//
// Algebraic simplification: scores @ (V ropeT) @ rope == scores @ V @ R with
// R = ropeT @ rope (64x64 constant). R is folded into Wo on device:
//   Wo' = blockdiag(R) @ Wo ;  y = attn_concat @ Wo' + bo
// => 4 dense GEMMs (4096x1024x1024) + flash attention (T=2048, hd=64),
// all on v_wmma_f32_16x16x32_bf16, f32 accumulate.
// Tile fills use gfx1250 GLOBAL_LOAD_ASYNC_TO_LDS_B128 (ASYNCcnt), double-
// buffered: s_wait_asynccnt N overlaps next-tile DMA with current WMMAs.
// ---------------------------------------------------------------------------

#define B_   2
#define T_   2048
#define D_   1024
#define H_   16
#define HD_  64
#define MTOT (B_ * T_)   // 4096

typedef __bf16 bf16_t;
typedef __attribute__((ext_vector_type(16))) __bf16 v16bf;
typedef __attribute__((ext_vector_type(8)))  __bf16 v8bf;
typedef __attribute__((ext_vector_type(8)))  float  v8f;

union FragU { v16bf v; v8bf h[2]; };

__device__ __forceinline__ bf16_t f2bf(float f) {
  unsigned u = __builtin_bit_cast(unsigned, f);
  unsigned r = (u + 0x7FFFu + ((u >> 16) & 1u)) >> 16;   // round-to-nearest-even
  unsigned short s = (unsigned short)r;
  return __builtin_bit_cast(bf16_t, s);
}

__device__ __forceinline__ v8f vzero8() {
  v8f z;
#pragma unroll
  for (int i = 0; i < 8; ++i) z[i] = 0.f;
  return z;
}

// gfx1250 async DMA: 16B per lane, global -> LDS, tracked by ASYNCcnt.
// Generic pointers to LDS carry the LDS byte offset in their low 32 bits.
__device__ __forceinline__ void async_load_b128(void* lds_dst, const void* gsrc) {
  unsigned lds = (unsigned)(unsigned long long)(uintptr_t)lds_dst;
  unsigned long long ga = (unsigned long long)(uintptr_t)gsrc;
  asm volatile("global_load_async_to_lds_b128 %0, %1, off"
               :: "v"(lds), "v"(ga)
               : "memory");
}
// Async loads complete in order per wave -> waiting to <=N outstanding
// guarantees all but the newest N have landed (pipelined double buffering).
__device__ __forceinline__ void wait_async0() {
  asm volatile("s_wait_asynccnt 0x0" ::: "memory");
}
__device__ __forceinline__ void wait_async4() {
  asm volatile("s_wait_asynccnt 0x4" ::: "memory");
}
__device__ __forceinline__ void wait_async8() {
  asm volatile("s_wait_asynccnt 0x8" ::: "memory");
}

// Load one 16x32 bf16 WMMA operand fragment (A or B layout; identical striping)
// from LDS at row-major [16][stride]. Per CDNA5 ISA 7.12.2:
// lanes 0-15 hold K 0..7 & 16..23, lanes 16-31 hold K 8..15 & 24..31.
__device__ __forceinline__ v16bf ld_frag(const bf16_t* p, int stride) {
  const int lane = threadIdx.x & 31;
  const bf16_t* q = p + (lane & 15) * stride + ((lane < 16) ? 0 : 8);
  FragU u;
  u.h[0] = *(const v8bf*)(q);        // K kb .. kb+7
  u.h[1] = *(const v8bf*)(q + 16);   // K kb+16 .. kb+23
  return u.v;
}

__device__ __forceinline__ v8f wmma_bf16(v16bf a, v16bf b, v8f c) {
  return __builtin_amdgcn_wmma_f32_16x16x32_bf16(
      /*neg_a=*/false, a, /*neg_b=*/false, b,
      /*c_mod=*/(short)0, c, /*reuse_a=*/false, /*reuse_b=*/false);
}

// ---------------------------------------------------------------------------
// Prep kernels
// ---------------------------------------------------------------------------

__global__ __launch_bounds__(256) void convert_f32_to_bf16_kernel(
    const float* __restrict__ in, bf16_t* __restrict__ out, int n) {
  int i = (blockIdx.x * 256 + threadIdx.x) * 4;
  if (i + 3 < n) {
    float4 v = *(const float4*)(in + i);
    out[i + 0] = f2bf(v.x);
    out[i + 1] = f2bf(v.y);
    out[i + 2] = f2bf(v.z);
    out[i + 3] = f2bf(v.w);
  }
}

// W (D x D, row-major) -> Wt = W^T as bf16 (N-major rows of length K).
__global__ void convert_wT_kernel(const float* __restrict__ W,
                                  bf16_t* __restrict__ Wt) {
  __shared__ float tile[32][33];
  const int tx = threadIdx.x, ty = threadIdx.y;
  const int bx = blockIdx.x, by = blockIdx.y;
#pragma unroll
  for (int i = 0; i < 4; ++i)
    tile[ty + i * 8][tx] = W[(size_t)(by * 32 + ty + i * 8) * D_ + bx * 32 + tx];
  __syncthreads();
#pragma unroll
  for (int i = 0; i < 4; ++i)
    Wt[(size_t)(bx * 32 + ty + i * 8) * D_ + by * 32 + tx] =
        f2bf(tile[tx][ty + i * 8]);
}

// R[i][j] = sum_s rope[s][i] * rope[s][j], rope computed analytically. 1 block.
__global__ __launch_bounds__(256) void rope_R_kernel(float* __restrict__ R) {
  __shared__ float rs[64];
  const int tid = threadIdx.x;
  const int i  = tid >> 2;          // 0..63 : row of R
  const int jb = (tid & 3) * 16;    // column base
  float acc[16];
#pragma unroll
  for (int e = 0; e < 16; ++e) acc[e] = 0.f;
  const float invf = __powf(10000.f, -(float)(tid & 31) / 32.f);
  for (int s = 0; s < T_; ++s) {
    __syncthreads();
    if (tid < 64) {
      float a = (float)s * invf;
      rs[tid] = (tid < 32) ? __cosf(a) : __sinf(a);
    }
    __syncthreads();
    float ri = rs[i];
#pragma unroll
    for (int e = 0; e < 16; ++e) acc[e] += ri * rs[jb + e];
  }
#pragma unroll
  for (int e = 0; e < 16; ++e) R[i * 64 + jb + e] = acc[e];
}

// WoT[j][i] = (blockdiag(R) @ Wo)^T in bf16:  Wo'[i][j] = sum_t R[i%64][t]*Wo[(i&~63)+t][j]
__global__ __launch_bounds__(256) void fold_wo_kernel(
    const float* __restrict__ R, const float* __restrict__ Wo,
    bf16_t* __restrict__ WoT) {
  const int i = blockIdx.x;           // 0..1023 : row of Wo'
  const int ih = i & 63;
  const int base = i & ~63;
  __shared__ float Rrow[64];
  if (threadIdx.x < 64) Rrow[threadIdx.x] = R[ih * 64 + threadIdx.x];
  __syncthreads();
  for (int j = threadIdx.x; j < D_; j += 256) {
    float acc = 0.f;
#pragma unroll 8
    for (int t = 0; t < 64; ++t)
      acc += Rrow[t] * Wo[(size_t)(base + t) * D_ + j];
    WoT[(size_t)j * D_ + i] = f2bf(acc);
  }
}

// ---------------------------------------------------------------------------
// NT GEMM: C(MxN) = alpha * A(MxK) * Bt(NxK)^T  [+ bias], bf16 in, WMMA.
// MODE 0: f32 out + bias   MODE 1: bf16 out   MODE 2: bf16 out, V^T head layout
// Block tile 128x128, K-step 64, 8 waves (2x4), 16 WMMA/wave per refill,
// double-buffered async-to-LDS tile pipeline.
// ---------------------------------------------------------------------------
#define BMG 128
#define BNG 128
#define BKG 64
#define LDSK 72   // padded stride: 144B rows -> conflict-free b128 frag reads

template <int MODE>
__global__ __launch_bounds__(256) void gemm_nt(
    const bf16_t* __restrict__ A, const bf16_t* __restrict__ Bt,
    void* __restrict__ Cout, const float* __restrict__ bias,
    float alpha, int M, int N, int K) {
  __shared__ __attribute__((aligned(16))) bf16_t As[2][BMG][LDSK];
  __shared__ __attribute__((aligned(16))) bf16_t Bs[2][BNG][LDSK];

  const int tid  = threadIdx.x;
  const int lane = tid & 31;
  const int wave = tid >> 5;
  const int wm = wave >> 2;            // 0..1  -> 64 rows each
  const int wn = wave & 3;             // 0..3  -> 32 cols each
  const int m0 = blockIdx.y * BMG;
  const int n0 = blockIdx.x * BNG;

  v8f acc[4][2];
#pragma unroll
  for (int mt = 0; mt < 4; ++mt)
#pragma unroll
    for (int nt = 0; nt < 2; ++nt) acc[mt][nt] = vzero8();

  const bf16_t* Ag = A  + (size_t)m0 * K;
  const bf16_t* Bg = Bt + (size_t)n0 * K;
  const int NIT = K / BKG;

  // prologue: DMA tile 0 into buffer 0
#pragma unroll
  for (int q = 0; q < 4; ++q) {
    int idx = tid * 8 + q * 2048;
    int r = idx >> 6, c = idx & 63;
    async_load_b128(&As[0][r][c], Ag + (size_t)r * K + c);
    async_load_b128(&Bs[0][r][c], Bg + (size_t)r * K + c);
  }

  for (int it = 0; it < NIT; ++it) {
    const int cur = it & 1;
    if (it + 1 < NIT) {
      const int k1 = (it + 1) * BKG;
      // issue next tile into the other buffer, then wait only for current tile
#pragma unroll
      for (int q = 0; q < 4; ++q) {
        int idx = tid * 8 + q * 2048;
        int r = idx >> 6, c = idx & 63;
        async_load_b128(&As[cur ^ 1][r][c], Ag + (size_t)r * K + k1 + c);
        async_load_b128(&Bs[cur ^ 1][r][c], Bg + (size_t)r * K + k1 + c);
      }
      wait_async8();   // 8 newest (next tile) may remain in flight
    } else {
      wait_async0();
    }
    __syncthreads();

#pragma unroll
    for (int kk = 0; kk < 2; ++kk) {
      v16bf fb0 = ld_frag(&Bs[cur][wn * 32][kk * 32], LDSK);
      v16bf fb1 = ld_frag(&Bs[cur][wn * 32 + 16][kk * 32], LDSK);
#pragma unroll
      for (int mt = 0; mt < 4; ++mt) {
        v16bf fa = ld_frag(&As[cur][wm * 64 + mt * 16][kk * 32], LDSK);
        acc[mt][0] = wmma_bf16(fa, fb0, acc[mt][0]);
        acc[mt][1] = wmma_bf16(fa, fb1, acc[mt][1]);
      }
    }
    __syncthreads();   // frees buffer `cur` for re-issue next iteration
  }

  const int cw = lane & 15;
  const int rb = (lane < 16) ? 0 : 8;
#pragma unroll
  for (int mt = 0; mt < 4; ++mt) {
#pragma unroll
    for (int nt = 0; nt < 2; ++nt) {
      const int n = n0 + wn * 32 + nt * 16 + cw;
#pragma unroll
      for (int r = 0; r < 8; ++r) {
        const int m = m0 + wm * 64 + mt * 16 + rb + r;
        const float v = acc[mt][nt][r] * alpha;
        if (MODE == 0) {
          ((float*)Cout)[(size_t)m * N + n] = v + bias[n];
        } else if (MODE == 1) {
          ((bf16_t*)Cout)[(size_t)m * N + n] = f2bf(v);
        } else {  // V^T head layout: [b][h][d][t]
          const int b = m >> 11, t = m & 2047;
          const int h = n >> 6,  d = n & 63;
          ((bf16_t*)Cout)[(((size_t)(b * H_ + h)) * HD_ + d) * T_ + t] = f2bf(v);
        }
      }
    }
  }
}

// ---------------------------------------------------------------------------
// Flash attention: Q,K in [B,T,D] bf16 (Q pre-scaled by 1/8), Vt in [B,H,64,T].
// Grid (T/128, B*H), 256 threads = 8 waves; wave w owns query rows w*16..+15.
// K/V tiles double-buffered via async-to-LDS + s_wait_asynccnt 4.
// ---------------------------------------------------------------------------
__global__ __launch_bounds__(256) void flash_attn_kernel(
    const bf16_t* __restrict__ Q, const bf16_t* __restrict__ K,
    const bf16_t* __restrict__ Vt, bf16_t* __restrict__ Out) {
  __shared__ __attribute__((aligned(16))) bf16_t Qs[128][72];
  __shared__ __attribute__((aligned(16))) bf16_t Ks[2][64][72];
  __shared__ __attribute__((aligned(16))) bf16_t Vs[2][64][72];   // [d][key]
  __shared__ __attribute__((aligned(16))) bf16_t Ps[8][16][72];   // per-wave P

  const int tid  = threadIdx.x;
  const int wave = tid >> 5;
  const int lane = tid & 31;
  const int cw = lane & 15;
  const int rb = (lane < 16) ? 0 : 8;
  const int bh = blockIdx.y;
  const int b = bh >> 4, h = bh & 15;
  const int q0 = blockIdx.x * 128;

  const bf16_t* Qg = Q + (size_t)(b * T_ + q0) * D_ + h * HD_;
  const bf16_t* Kg = K + (size_t)(b * T_) * D_ + h * HD_;
  const bf16_t* Vg = Vt + (size_t)bh * HD_ * T_;

  // async DMA the Q tile, then key/value tile 0 into buffer 0
#pragma unroll
  for (int p = 0; p < 4; ++p) {
    int idx = tid * 8 + p * 2048;
    int r = idx >> 6, c = idx & 63;
    async_load_b128(&Qs[r][c], Qg + (size_t)r * D_ + c);
  }
#pragma unroll
  for (int p = 0; p < 2; ++p) {
    int idx = tid * 8 + p * 2048;
    int r = idx >> 6, c = idx & 63;
    async_load_b128(&Ks[0][r][c], Kg + (size_t)r * D_ + c);
    async_load_b128(&Vs[0][r][c], Vg + (size_t)r * T_ + c);
  }

  v8f o[4];
#pragma unroll
  for (int j = 0; j < 4; ++j) o[j] = vzero8();
  float mstate[8], lstate[8];
#pragma unroll
  for (int r = 0; r < 8; ++r) { mstate[r] = -3.0e38f; lstate[r] = 0.f; }

  const int NB = T_ / 64;
  for (int nb = 0; nb < NB; ++nb) {
    const int cur = nb & 1;
    if (nb + 1 < NB) {
      const int t1 = (nb + 1) * 64;
#pragma unroll
      for (int p = 0; p < 2; ++p) {
        int idx = tid * 8 + p * 2048;
        int r = idx >> 6, c = idx & 63;
        async_load_b128(&Ks[cur ^ 1][r][c], Kg + (size_t)(t1 + r) * D_ + c);
        async_load_b128(&Vs[cur ^ 1][r][c], Vg + (size_t)r * T_ + t1 + c);
      }
      if (nb + 2 < NB) {  // light prefetch of tile nb+2 -> global_prefetch_b8
        int idx = tid * 8;
        int r = idx >> 6, c = idx & 63;
        __builtin_prefetch(Kg + (size_t)(t1 + 64 + r) * D_ + c, 0, 1);
        __builtin_prefetch(Vg + (size_t)r * T_ + t1 + 64 + c, 0, 1);
      }
      wait_async4();   // 4 newest (next tile) may remain in flight
    } else {
      wait_async0();
    }
    __syncthreads();

    // S = Q_tile . K_tile^T  (K contraction = 64 -> two WMMA steps)
    v16bf fa0 = ld_frag(&Qs[wave * 16][0], 72);
    v16bf fa1 = ld_frag(&Qs[wave * 16][32], 72);
    v8f s[4];
#pragma unroll
    for (int j = 0; j < 4; ++j) {
      s[j] = vzero8();
      v16bf fb0 = ld_frag(&Ks[cur][j * 16][0], 72);
      v16bf fb1 = ld_frag(&Ks[cur][j * 16][32], 72);
      s[j] = wmma_bf16(fa0, fb0, s[j]);
      s[j] = wmma_bf16(fa1, fb1, s[j]);
    }

    // online softmax (row = rb + r, spread over 16 lanes of each half-wave)
#pragma unroll
    for (int r = 0; r < 8; ++r) {
      float rmax = fmaxf(fmaxf(s[0][r], s[1][r]), fmaxf(s[2][r], s[3][r]));
#pragma unroll
      for (int off = 1; off < 16; off <<= 1)
        rmax = fmaxf(rmax, __shfl_xor(rmax, off, 32));
      const float mnew = fmaxf(mstate[r], rmax);
      const float corr = __expf(mstate[r] - mnew);
      mstate[r] = mnew;
      float rsum = 0.f;
#pragma unroll
      for (int j = 0; j < 4; ++j) {
        float pv = __expf(s[j][r] - mnew);
        s[j][r] = pv;
        rsum += pv;
      }
#pragma unroll
      for (int off = 1; off < 16; off <<= 1)
        rsum += __shfl_xor(rsum, off, 32);
      lstate[r] = lstate[r] * corr + rsum;
#pragma unroll
      for (int j = 0; j < 4; ++j) o[j][r] *= corr;
    }

    // P: C-fragment layout -> LDS -> A-fragment layout (per-wave scratch)
#pragma unroll
    for (int j = 0; j < 4; ++j)
#pragma unroll
      for (int r = 0; r < 8; ++r)
        Ps[wave][rb + r][j * 16 + cw] = f2bf(s[j][r]);

    v16bf fp0 = ld_frag(&Ps[wave][0][0], 72);
    v16bf fp1 = ld_frag(&Ps[wave][0][32], 72);
#pragma unroll
    for (int j = 0; j < 4; ++j) {
      v16bf fv0 = ld_frag(&Vs[cur][j * 16][0], 72);
      v16bf fv1 = ld_frag(&Vs[cur][j * 16][32], 72);
      o[j] = wmma_bf16(fp0, fv0, o[j]);
      o[j] = wmma_bf16(fp1, fv1, o[j]);
    }
    __syncthreads();   // frees buffer `cur` for re-issue next iteration
  }

  // normalize and write merged-head bf16 output [B,T,D]
  bf16_t* Og = Out + (size_t)(b * T_ + q0 + wave * 16) * D_ + h * HD_;
#pragma unroll
  for (int r = 0; r < 8; ++r) {
    const float inv = 1.f / lstate[r];
#pragma unroll
    for (int j = 0; j < 4; ++j)
      Og[(size_t)(rb + r) * D_ + j * 16 + cw] = f2bf(o[j][r] * inv);
  }
}

// ---------------------------------------------------------------------------
extern "C" void kernel_launch(void* const* d_in, const int* in_sizes, int n_in,
                              void* d_out, int out_size, void* d_ws,
                              size_t ws_size, hipStream_t stream) {
  (void)in_sizes; (void)n_in; (void)out_size; (void)ws_size;
  const float* x  = (const float*)d_in[0];
  const float* Wq = (const float*)d_in[1];
  const float* Wk = (const float*)d_in[2];
  const float* Wv = (const float*)d_in[3];
  const float* Wo = (const float*)d_in[4];
  const float* bo = (const float*)d_in[5];
  float* out = (float*)d_out;

  // workspace layout (~42 MB)
  char* ws = (char*)d_ws;
  float* R = (float*)ws;                        // 16 KB
  bf16_t* bws = (bf16_t*)(ws + 16384);
  const size_t XSZ = (size_t)MTOT * D_;         // 4M elems
  const size_t WSZ = (size_t)D_ * D_;           // 1M elems
  bf16_t* xb   = bws;
  bf16_t* Wqt  = xb + XSZ;
  bf16_t* Wkt  = Wqt + WSZ;
  bf16_t* Wvt  = Wkt + WSZ;
  bf16_t* WoT  = Wvt + WSZ;
  bf16_t* Qb   = WoT + WSZ;
  bf16_t* Kb   = Qb + XSZ;
  bf16_t* Vt   = Kb + XSZ;
  bf16_t* attn = xb;  // xb no longer needed after the three projections

  convert_f32_to_bf16_kernel<<<dim3((unsigned)(XSZ / 1024)), 256, 0, stream>>>(
      x, xb, (int)XSZ);
  dim3 tb(32, 8), tg(32, 32);
  convert_wT_kernel<<<tg, tb, 0, stream>>>(Wq, Wqt);
  convert_wT_kernel<<<tg, tb, 0, stream>>>(Wk, Wkt);
  convert_wT_kernel<<<tg, tb, 0, stream>>>(Wv, Wvt);
  rope_R_kernel<<<1, 256, 0, stream>>>(R);
  fold_wo_kernel<<<D_, 256, 0, stream>>>(R, Wo, WoT);

  dim3 gg(D_ / BNG, MTOT / BMG);  // (8, 32)
  // Q projection with softmax scale 1/sqrt(64) folded in.
  gemm_nt<1><<<gg, 256, 0, stream>>>(xb, Wqt, Qb, nullptr, 0.125f, MTOT, D_, D_);
  gemm_nt<1><<<gg, 256, 0, stream>>>(xb, Wkt, Kb, nullptr, 1.0f, MTOT, D_, D_);
  gemm_nt<2><<<gg, 256, 0, stream>>>(xb, Wvt, Vt, nullptr, 1.0f, MTOT, D_, D_);

  flash_attn_kernel<<<dim3(T_ / 128, B_ * H_), 256, 0, stream>>>(Qb, Kb, Vt, attn);

  gemm_nt<0><<<gg, 256, 0, stream>>>(attn, WoT, out, bo, 1.0f, MTOT, D_, D_);
}
